// ESM2ContactPredictor_15109694947951
// MI455X (gfx1250) — matrix operand order repaired and checked by reading, not compiled, not run on hardware
//
#include <hip/hip_runtime.h>
#include <hip/hip_bf16.h>

typedef __attribute__((ext_vector_type(16))) _Float16 v16h;
typedef __attribute__((ext_vector_type(8)))  float    v8f;

#define LL 512   // sequence length
#define DD 1280  // embed dim
#define H1 512   // hidden 1 (GEMM2 K)
#define H2 128   // hidden 2 (GEMM2 N)

// ---------------------------------------------------------------------------
// CDNA5 async global->LDS copy (ASYNCcnt-tracked, no VGPR data return).
// GVS addressing: mem = SGPR_base + VGPR_offset; VDST = LDS byte address.
// ---------------------------------------------------------------------------
__device__ __forceinline__ void async_ld_b128(uint32_t lds_addr, uint32_t voff,
                                              const void* sbase) {
  asm volatile("global_load_async_to_lds_b128 %0, %1, %2"
               :: "v"(lds_addr), "v"(voff), "s"(sbase)
               : "memory");
}
__device__ __forceinline__ void wait_async0() {
  asm volatile("s_wait_asynccnt 0x0" ::: "memory");
}

// ---------------------------------------------------------------------------
// Pack an f32 weight matrix W[K x N] (row-major) into f16 WMMA B fragments for
// v_wmma_f32_16x16x32_f16 (wave32). B is 32(K) x 16(N) per fragment:
//   lane <16 : column n = nt*16 + lane,     K = ks*32 + {0..7, 16..23}
//   lane>=16 : column n = nt*16 + lane-16,  K = ks*32 + {8..15, 24..31}
// flat element index: ((ks*ntiles + nt)*32 + lane)*16 + t   (16 halfs = 32B/lane)
// ---------------------------------------------------------------------------
__global__ __launch_bounds__(256) void pack_b_kernel(
    const float* __restrict__ W, _Float16* __restrict__ outh,
    int ksteps, int ntiles, int N, int total)
{
  int id = blockIdx.x * 256 + threadIdx.x;
  if (id >= total) return;
  int t    = id & 15;
  int lane = (id >> 4) & 31;
  int fi   = id >> 9;
  int nt   = fi % ntiles;
  int ks   = fi / ntiles;
  int k = ks * 32 + ((lane < 16) ? 0 : 8) + ((t < 8) ? t : t + 8);
  int n = nt * 16 + (lane & 15);
  outh[id] = (_Float16)W[(size_t)k * N + n];
}

// ---------------------------------------------------------------------------
// GEMM1: xi = emb @ W1[:DD], xj = emb @ W1[DD:]   (f32 out, f16 WMMA compute)
// W1 packed as K=2560 (80 ksteps), N=512 (32 ntiles).
// One wave: 16 rows x 64 cols, K=1280 (40 ksteps). 512 wave-tasks = 64 blocks.
// A fragment layout (16-bit A 16x32): lane<16: K {0..7, 16..23}; lane>=16:
// K {8..15, 24..31}; row M = lane&15.
// ---------------------------------------------------------------------------
__global__ __launch_bounds__(256) void gemm1_kernel(
    const float* __restrict__ emb,
    const v16h* __restrict__ w1h,
    float* __restrict__ xi, float* __restrict__ xj)
{
  const int wave = threadIdx.x >> 5;
  const int lane = threadIdx.x & 31;
  const int task = blockIdx.x * 8 + wave;      // 0..511
  const int half = task >> 8;                  // 0: xi, 1: xj
  const int rem  = task & 255;
  const int mt   = rem >> 3;                   // 0..31 row tile
  const int cc   = rem & 7;                    // 0..7 col chunk of 64
  const int i0   = mt * 16;
  const int m    = lane & 15;
  const int koff = (lane < 16) ? 0 : 8;
  const float* arow = emb + (size_t)(i0 + m) * DD;

  // loop-invariant fragment base: per-ks stride = 32 ntiles * 32 lanes = 1024
  const v16h* bbase = w1h + ((size_t)(half * 40) * 32 + cc * 4) * 32 + lane;

  v8f acc[4] = {};
  for (int ks = 0; ks < 40; ++ks) {
    // issue all 4 B-fragment loads first (latency hides under A build)
    v16h bfrag[4];
#pragma unroll
    for (int t = 0; t < 4; ++t) bfrag[t] = bbase[ks * 1024 + t * 32];

    const int kb = ks * 32 + koff;
    v16h a;
#pragma unroll
    for (int t = 0; t < 8; ++t) {
      a[t]     = (_Float16)arow[kb + t];
      a[8 + t] = (_Float16)arow[kb + 16 + t];
    }
#pragma unroll
    for (int t = 0; t < 4; ++t)
      acc[t] = __builtin_amdgcn_wmma_f32_16x16x32_f16(
          false, a, false, bfrag[t], (short)0, acc[t], false, false);
  }
  // C/D layout: lane<16: N=lane, M=r; lane>=16: N=lane-16, M=8+r
  float* dst = half ? xj : xi;
  const int mbase = (lane < 16) ? 0 : 8;
#pragma unroll
  for (int t = 0; t < 4; ++t)
#pragma unroll
    for (int r = 0; r < 8; ++r)
      dst[(size_t)(i0 + mbase + r) * H1 + cc * 64 + t * 16 + m] = acc[t][r];
}

// ---------------------------------------------------------------------------
// Fused main kernel: for every pair (i,j):
//   h  = relu(xi[i] + xj[j] + b1)          [512]
//   h2 = relu(h @ W2 + b2)                 [128]
//   out[i,j] = sigmoid(h2 . W3 + b3)
// Block = 8 waves: 8 consecutive i sharing one 16-wide j tile (xj in LDS).
// Wave = one i x 16 j (GEMM M=16) x full N=128 (8 accumulators), K=512.
// LDS staged with CDNA5 async global->LDS B128 copies (ASYNCcnt).
// ---------------------------------------------------------------------------
__global__ __launch_bounds__(256) void fused_contact_kernel(
    const float* __restrict__ xi, const float* __restrict__ xj,
    const float* __restrict__ b1, const v16h* __restrict__ w2h,
    const float* __restrict__ b2, const float* __restrict__ W3,
    const float* __restrict__ b3, float* __restrict__ out)
{
  __shared__ float s_xj[16 * H1];  // 32 KB: xj rows j0..j0+15
  __shared__ float s_xi[8 * H1];   // 16 KB: xi rows for the 8 waves
  __shared__ float s_b1[H1];       //  2 KB

  const int wave = threadIdx.x >> 5;
  const int lane = threadIdx.x & 31;
  const int jt   = blockIdx.x & 31;    // j tile  (32)
  const int ig   = blockIdx.x >> 5;    // i group (64)
  const int j0   = jt * 16;
  const int i    = ig * 8 + wave;

  // per-ks fragment stride = 8 ntiles * 32 lanes = 256 v16h (8 KB)
  const v16h* bbase = w2h + lane;
  __builtin_prefetch(bbase, 0, 3);     // warm WGP$/L2 during staging

  // --- async stage xj tile / xi rows / b1 into LDS (16B per lane-op) ---
  {
    const uint32_t l_xj = (uint32_t)(uintptr_t)s_xj;
    const uint32_t l_xi = (uint32_t)(uintptr_t)s_xi;
    const uint32_t l_b1 = (uint32_t)(uintptr_t)s_b1;
    const float* g_xj = xj + (size_t)j0 * H1;
    const float* g_xi = xi + (size_t)(ig * 8) * H1;
#pragma unroll
    for (int t = threadIdx.x; t < 16 * H1 / 4; t += 256)   // 8 iters
      async_ld_b128(l_xj + t * 16, (uint32_t)(t * 16), g_xj);
#pragma unroll
    for (int t = threadIdx.x; t < 8 * H1 / 4; t += 256)    // 4 iters
      async_ld_b128(l_xi + t * 16, (uint32_t)(t * 16), g_xi);
    if (threadIdx.x < H1 / 4)                              // 128 lane-ops
      async_ld_b128(l_b1 + threadIdx.x * 16, (uint32_t)(threadIdx.x * 16), b1);
    wait_async0();
  }
  __syncthreads();

  v8f acc[8] = {};
  const int  koff  = (lane < 16) ? 0 : 8;
  const int  m     = lane & 15;                 // pair row within wave tile (j offset)
  const float* xjrow = s_xj + m * H1;
  const float* xirow = s_xi + wave * H1;

  for (int ks = 0; ks < 16; ++ks) {
    // issue all 8 B-fragment loads up front; latency overlaps the A build
    v16h bfrag[8];
#pragma unroll
    for (int t = 0; t < 8; ++t) bfrag[t] = bbase[ks * 256 + t * 32];

    const int kb = ks * 32 + koff;
    v16h a;
#pragma unroll
    for (int t = 0; t < 8; ++t) {
      float v0 = xirow[kb + t]      + xjrow[kb + t]      + s_b1[kb + t];
      float v1 = xirow[kb + 16 + t] + xjrow[kb + 16 + t] + s_b1[kb + 16 + t];
      a[t]     = (_Float16)fmaxf(v0, 0.0f);
      a[8 + t] = (_Float16)fmaxf(v1, 0.0f);
    }
#pragma unroll
    for (int t = 0; t < 8; ++t)
      acc[t] = __builtin_amdgcn_wmma_f32_16x16x32_f16(
          false, a, false, bfrag[t], (short)0, acc[t], false, false);
  }

  // Epilogue: h2 = relu(acc + b2), partial[r] += h2 * W3[n]
  const float b3v = b3[0];
  float part[8];
#pragma unroll
  for (int r = 0; r < 8; ++r) part[r] = 0.0f;
#pragma unroll
  for (int t = 0; t < 8; ++t) {
    const int n   = t * 16 + m;
    const float w3v = W3[n];
    const float b2v = b2[n];
#pragma unroll
    for (int r = 0; r < 8; ++r) {
      float h2 = fmaxf(acc[t][r] + b2v, 0.0f);
      part[r] += h2 * w3v;
    }
  }
  // reduce over the 16 lanes sharing the same C rows (xor stays inside half-wave)
#pragma unroll
  for (int r = 0; r < 8; ++r) {
    float v = part[r];
    v += __shfl_xor(v, 1, 32);
    v += __shfl_xor(v, 2, 32);
    v += __shfl_xor(v, 4, 32);
    v += __shfl_xor(v, 8, 32);
    part[r] = v;
  }
  // lane 0 holds rows 0..7, lane 16 holds rows 8..15; 8 consecutive j's per
  // lane -> two 16B vector stores
  if ((lane & 15) == 0) {
    const int rbase = (lane >> 4) * 8;
    float4 o0, o1;
    o0.x = 1.0f / (1.0f + __expf(-(part[0] + b3v)));
    o0.y = 1.0f / (1.0f + __expf(-(part[1] + b3v)));
    o0.z = 1.0f / (1.0f + __expf(-(part[2] + b3v)));
    o0.w = 1.0f / (1.0f + __expf(-(part[3] + b3v)));
    o1.x = 1.0f / (1.0f + __expf(-(part[4] + b3v)));
    o1.y = 1.0f / (1.0f + __expf(-(part[5] + b3v)));
    o1.z = 1.0f / (1.0f + __expf(-(part[6] + b3v)));
    o1.w = 1.0f / (1.0f + __expf(-(part[7] + b3v)));
    float4* dst = (float4*)(out + (size_t)i * LL + j0 + rbase);
    dst[0] = o0;
    dst[1] = o1;
  }
}

// ---------------------------------------------------------------------------
// Workspace layout (bytes):
//   [0)                 w1h: 80*32*32*16 f16 = 2,621,440
//   [2,621,440)         w2h: 16* 8*32*16 f16 =   131,072
//   [2,752,512)         xi : 512*512 f32     = 1,048,576
//   [3,801,088)         xj : 512*512 f32     = 1,048,576     (end 4,849,664)
// ---------------------------------------------------------------------------
extern "C" void kernel_launch(void* const* d_in, const int* in_sizes, int n_in,
                              void* d_out, int out_size, void* d_ws, size_t ws_size,
                              hipStream_t stream) {
  const float* emb = (const float*)d_in[0];
  const float* W1  = (const float*)d_in[1];
  const float* b1  = (const float*)d_in[2];
  const float* W2  = (const float*)d_in[3];
  const float* b2  = (const float*)d_in[4];
  const float* W3  = (const float*)d_in[5];
  const float* b3  = (const float*)d_in[6];
  float* out = (float*)d_out;

  char* ws = (char*)d_ws;
  _Float16* w1h = (_Float16*)(ws);
  _Float16* w2h = (_Float16*)(ws + 2621440);
  float*    xi  = (float*)(ws + 2752512);
  float*    xj  = (float*)(ws + 3801088);

  const int totW1 = 80 * 32 * 32 * 16;
  pack_b_kernel<<<(totW1 + 255) / 256, 256, 0, stream>>>(W1, w1h, 80, 32, 512, totW1);
  const int totW2 = 16 * 8 * 32 * 16;
  pack_b_kernel<<<(totW2 + 255) / 256, 256, 0, stream>>>(W2, w2h, 16, 8, 128, totW2);

  gemm1_kernel<<<64, 256, 0, stream>>>(emb, (const v16h*)w1h, xi, xj);

  fused_contact_kernel<<<2048, 256, 0, stream>>>(xi, xj, b1, (const v16h*)w2h,
                                                 b2, W3, b3, out);
}
